// CausalSelfAttention_60344290509232
// MI455X (gfx1250) — matrix operand order repaired and checked by reading, not compiled, not run on hardware
//
#include <hip/hip_runtime.h>

typedef __bf16 bf16_t;
typedef __attribute__((ext_vector_type(16))) __bf16 v16bf;
typedef __attribute__((ext_vector_type(8)))  __bf16 v8bf;
typedef __attribute__((ext_vector_type(4)))  __bf16 v4bf;
typedef __attribute__((ext_vector_type(8)))  float  v8f;

#define B_SZ  2
#define T_SZ  2048
#define C_SZ  1024
#define NH    16
#define HD    64
#define HEAD_ELEMS (B_SZ * NH * T_SZ * HD)   // 4194304 bf16 elems per Q/K/V buffer

// ---------------------------------------------------------------------------
// WMMA helpers: v_wmma_f32_16x16x32_bf16, D = A(16x32) * B(32x16) + C(16x16 f32)
// A layout (16-bit): lane row = lane&15; elems 0..7 -> K=(lane>>4)*8+e,
//                    elems 8..15 -> K=16+(lane>>4)*8+(e-8)
// B layout (16-bit): lane col = lane&15; same K mapping as A
// C/D layout (f32):  lane col = lane&15; vgpr v -> row M=(lane>>4)*8+v
// ---------------------------------------------------------------------------
__device__ __forceinline__ v8f wmma_bf16(v16bf a, v16bf b, v8f c) {
  return __builtin_amdgcn_wmma_f32_16x16x32_bf16(
      /*neg_a=*/false, a, /*neg_b=*/false, b,
      /*c_mod=*/(short)0, c, /*reuse_a=*/false, /*reuse_b=*/false);
}

// Load an A fragment from a row-major [row][k] bf16 tile (ldk elements/row).
__device__ __forceinline__ v16bf load_frag_a(const bf16_t* tile, int ldk) {
  const int lane = threadIdx.x & 31;
  const int row  = lane & 15;
  const int kb   = (lane >> 4) << 3;  // 0 or 8
  const bf16_t* p = tile + (size_t)row * ldk + kb;
  v8bf lo = *reinterpret_cast<const v8bf*>(p);
  v8bf hi = *reinterpret_cast<const v8bf*>(p + 16);
  v16bf r;
#pragma unroll
  for (int i = 0; i < 8; ++i) { r[i] = lo[i]; r[i + 8] = hi[i]; }
  return r;
}

// Load a B fragment from an [n][k] (transposed) bf16 tile (ldk elements/row).
__device__ __forceinline__ v16bf load_frag_b(const bf16_t* tile, int ldk) {
  const int lane = threadIdx.x & 31;
  const int col  = lane & 15;
  const int kb   = (lane >> 4) << 3;  // 0 or 8
  const bf16_t* p = tile + (size_t)col * ldk + kb;
  v8bf lo = *reinterpret_cast<const v8bf*>(p);
  v8bf hi = *reinterpret_cast<const v8bf*>(p + 16);
  v16bf r;
#pragma unroll
  for (int i = 0; i < 8; ++i) { r[i] = lo[i]; r[i + 8] = hi[i]; }
  return r;
}

__device__ __forceinline__ float half16_max(float v) {
  v = fmaxf(v, __shfl_xor(v, 1, 32));
  v = fmaxf(v, __shfl_xor(v, 2, 32));
  v = fmaxf(v, __shfl_xor(v, 4, 32));
  v = fmaxf(v, __shfl_xor(v, 8, 32));
  return v;
}
__device__ __forceinline__ float half16_sum(float v) {
  v += __shfl_xor(v, 1, 32);
  v += __shfl_xor(v, 2, 32);
  v += __shfl_xor(v, 4, 32);
  v += __shfl_xor(v, 8, 32);
  return v;
}

// ===========================================================================
// Kernel 1: qkv = x @ w_attn + b_attn, scatter to bf16 Q/K/V [B,H,T,HD]
// M=4096, N=3072, K=1024.  Block tile 128x128, 8 waves, wave tile 64x32.
// Software-pipelined: global->regs for step k+1 issued before WMMA on step k,
// regs->LDS (other buffer) after compute; one barrier per K-step.
// ===========================================================================
__global__ __launch_bounds__(256) void qkv_gemm_kernel(
    const float* __restrict__ x, const float* __restrict__ w_attn,
    const float* __restrict__ b_attn, unsigned short* __restrict__ qkv_ws) {
  __shared__ __align__(16) bf16_t Xs[2][128][32];
  __shared__ __align__(16) bf16_t Ws[2][128][32];  // [n][k] transposed

  bf16_t* Qb = reinterpret_cast<bf16_t*>(qkv_ws);
  bf16_t* Kb = Qb + HEAD_ELEMS;
  bf16_t* Vb = Qb + 2 * HEAD_ELEMS;

  const int tid  = threadIdx.x;
  const int wave = tid >> 5;
  const int lane = tid & 31;
  const int bN0  = blockIdx.x * 128;
  const int bM0  = blockIdx.y * 128;
  const int wM   = (wave >> 2) * 64;  // 0 / 64
  const int wN   = (wave & 3) * 32;   // 0..96

  v8f acc[4][2];
#pragma unroll
  for (int i = 0; i < 4; ++i)
#pragma unroll
    for (int j = 0; j < 2; ++j) acc[i][j] = {};

  float4 xreg[4], wreg[4];
  auto gload = [&](int k0) {
#pragma unroll
    for (int r = 0; r < 4; ++r) {
      const int i = tid + r * 256;            // 0..1023 (128x32 / 4)
      const int m = i >> 3, k4 = (i & 7) << 2;
      xreg[r] = *reinterpret_cast<const float4*>(
          x + (size_t)(bM0 + m) * C_SZ + k0 + k4);
    }
#pragma unroll
    for (int r = 0; r < 4; ++r) {
      const int i = tid + r * 256;
      const int k = i >> 5, n4 = (i & 31) << 2;
      wreg[r] = *reinterpret_cast<const float4*>(
          w_attn + (size_t)(k0 + k) * (3 * C_SZ) + bN0 + n4);
    }
  };
  auto lstore = [&](int buf) {
#pragma unroll
    for (int r = 0; r < 4; ++r) {
      const int i = tid + r * 256;
      const int m = i >> 3, k4 = (i & 7) << 2;
      v4bf o;
      o[0] = (bf16_t)xreg[r].x; o[1] = (bf16_t)xreg[r].y;
      o[2] = (bf16_t)xreg[r].z; o[3] = (bf16_t)xreg[r].w;
      *reinterpret_cast<v4bf*>(&Xs[buf][m][k4]) = o;
    }
#pragma unroll
    for (int r = 0; r < 4; ++r) {
      const int i = tid + r * 256;
      const int k = i >> 5, n4 = (i & 31) << 2;
      Ws[buf][n4 + 0][k] = (bf16_t)wreg[r].x;
      Ws[buf][n4 + 1][k] = (bf16_t)wreg[r].y;
      Ws[buf][n4 + 2][k] = (bf16_t)wreg[r].z;
      Ws[buf][n4 + 3][k] = (bf16_t)wreg[r].w;
    }
  };

  const int NKT = C_SZ / 32;
  gload(0);
  lstore(0);
  for (int kt = 0; kt < NKT; ++kt) {
    __syncthreads();
    const int buf = kt & 1;
    if (kt + 1 < NKT) gload((kt + 1) * 32);   // prefetch into regs (VMEM only)

    v16bf af[4];
#pragma unroll
    for (int tm = 0; tm < 4; ++tm)
      af[tm] = load_frag_a(&Xs[buf][wM + tm * 16][0], 32);
#pragma unroll
    for (int tn = 0; tn < 2; ++tn) {
      v16bf bfr = load_frag_b(&Ws[buf][wN + tn * 16][0], 32);
#pragma unroll
      for (int tm = 0; tm < 4; ++tm)
        acc[tm][tn] = wmma_bf16(af[tm], bfr, acc[tm][tn]);
    }
    if (kt + 1 < NKT) lstore(buf ^ 1);        // drain regs into other buffer
  }

  // Scatter store with bias: col n -> (which, h, d); row m -> (b, t)
#pragma unroll
  for (int tm = 0; tm < 4; ++tm) {
#pragma unroll
    for (int tn = 0; tn < 2; ++tn) {
      const int n     = bN0 + wN + tn * 16 + (lane & 15);
      const int mbase = bM0 + wM + tm * 16 + ((lane >> 4) << 3);
      const float bias = b_attn[n];
      const int which = n >> 10;
      const int c     = n & 1023;
      const int h     = c >> 6;
      const int d     = c & 63;
      bf16_t* dst = (which == 0) ? Qb : (which == 1 ? Kb : Vb);
#pragma unroll
      for (int v = 0; v < 8; ++v) {
        const int m = mbase + v;
        const int b = m >> 11;
        const int t = m & (T_SZ - 1);
        const size_t idx = ((((size_t)b << 4) + h) * T_SZ + t) * HD + d;
        dst[idx] = (bf16_t)(acc[tm][tn][v] + bias);
      }
    }
  }
}

// ===========================================================================
// Kernel 2: causal flash attention per (b,h). Grid (T/64, B*NH), 4 waves.
// Each wave: 16 q-rows, Q strip in registers, loop over 32-key chunks with
// double-buffered K/V staging (global->regs before compute, regs->LDS after).
// ===========================================================================
__global__ __launch_bounds__(128) void attn_kernel(
    const unsigned short* __restrict__ qkv_ws, unsigned short* __restrict__ y_ws) {
  __shared__ __align__(16) bf16_t Ks[2][32][64];    // [key][d]
  __shared__ __align__(16) bf16_t Vt[2][64][32];    // [d][key] (transposed)
  __shared__ __align__(16) bf16_t Ps[4][16][32];    // per-wave P staging

  const bf16_t* Qb = reinterpret_cast<const bf16_t*>(qkv_ws);
  const bf16_t* Kb = Qb + HEAD_ELEMS;
  const bf16_t* Vb = Qb + 2 * HEAD_ELEMS;
  bf16_t* Yb = reinterpret_cast<bf16_t*>(y_ws);

  const int tid  = threadIdx.x;
  const int wave = tid >> 5;
  const int lane = tid & 31;
  const int q0   = blockIdx.x * 64;
  const int bh   = blockIdx.y;        // b*NH + h
  const int b    = bh >> 4;
  const int h    = bh & 15;

  const size_t headoff = (size_t)bh * T_SZ * HD;
  const bf16_t* Qh = Qb + headoff;
  const bf16_t* Kh = Kb + headoff;
  const bf16_t* Vh = Vb + headoff;

  // Q strip (16 rows x 64) in registers as two A-frags (K=0..31, 32..63)
  v16bf qa0 = load_frag_a(Qh + (size_t)(q0 + wave * 16) * HD, HD);
  v16bf qa1 = load_frag_a(Qh + (size_t)(q0 + wave * 16) * HD + 32, HD);

  v8f o[4];
#pragma unroll
  for (int i = 0; i < 4; ++i) o[i] = {};
  float mrow[8], lrow[8];
#pragma unroll
  for (int v = 0; v < 8; ++v) { mrow[v] = -1.0e30f; lrow[v] = 0.0f; }

  const int rowbase = q0 + wave * 16 + ((lane >> 4) << 3);
  const int ncol    = lane & 15;
  const int nchunks = (q0 >> 5) + 2;  // keys 0 .. q0+63

  v8bf kreg[2], vreg[2];
  auto gloadKV = [&](int j0) {
#pragma unroll
    for (int r = 0; r < 2; ++r) {
      const int i = tid + r * 128;              // 0..255 (32 keys x 8 chunks)
      const int key = i >> 3, d8 = (i & 7) << 3;
      kreg[r] = *reinterpret_cast<const v8bf*>(&Kh[(size_t)(j0 + key) * HD + d8]);
      vreg[r] = *reinterpret_cast<const v8bf*>(&Vh[(size_t)(j0 + key) * HD + d8]);
    }
  };
  auto lstoreKV = [&](int buf) {
#pragma unroll
    for (int r = 0; r < 2; ++r) {
      const int i = tid + r * 128;
      const int key = i >> 3, d8 = (i & 7) << 3;
      *reinterpret_cast<v8bf*>(&Ks[buf][key][d8]) = kreg[r];
#pragma unroll
      for (int e = 0; e < 8; ++e) Vt[buf][d8 + e][key] = vreg[r][e];
    }
  };

  gloadKV(0);
  lstoreKV(0);
  for (int ch = 0; ch < nchunks; ++ch) {
    const int j0 = ch << 5;
    __syncthreads();
    const int buf = ch & 1;
    if (ch + 1 < nchunks) gloadKV((ch + 1) << 5);

    // S = Q @ K^T for two 16-key subtiles (scale + causal mask)
    v8f s[2];
#pragma unroll
    for (int jt = 0; jt < 2; ++jt) {
      v8f t = {};
      v16bf kb0 = load_frag_b(&Ks[buf][jt * 16][0], HD);       // d = 0..31
      v16bf kb1 = load_frag_b(&Ks[buf][jt * 16][0] + 32, HD);  // d = 32..63
      t = wmma_bf16(qa0, kb0, t);
      t = wmma_bf16(qa1, kb1, t);
#pragma unroll
      for (int v = 0; v < 8; ++v) {
        const int key = j0 + jt * 16 + ncol;
        const int row = rowbase + v;
        float val = t[v] * 0.125f;  // 1/sqrt(64)
        if (key > row) val = -1.0e30f;
        s[jt][v] = val;
      }
    }

    // Online softmax per row (row spans 16 lanes within each half-wave)
#pragma unroll
    for (int v = 0; v < 8; ++v) {
      const int row  = rowbase + v;
      const float ml = half16_max(fmaxf(s[0][v], s[1][v]));
      const float mn = fmaxf(mrow[v], ml);
      const float alpha = __expf(mrow[v] - mn);
      const float p0 = ((j0 + ncol) > row)      ? 0.0f : __expf(s[0][v] - mn);
      const float p1 = ((j0 + 16 + ncol) > row) ? 0.0f : __expf(s[1][v] - mn);
      lrow[v] = lrow[v] * alpha + half16_sum(p0 + p1);
      mrow[v] = mn;
#pragma unroll
      for (int tn = 0; tn < 4; ++tn) o[tn][v] *= alpha;
      const int rl = ((lane >> 4) << 3) + v;  // local row in strip
      Ps[wave][rl][ncol]      = (bf16_t)p0;
      Ps[wave][rl][16 + ncol] = (bf16_t)p1;
    }

    // O += P(16x32) @ V(32x64); LDS is in-order per wave, Ps is wave-private
    v16bf pa = load_frag_a(&Ps[wave][0][0], 32);
#pragma unroll
    for (int tn = 0; tn < 4; ++tn) {
      v16bf vb = load_frag_b(&Vt[buf][tn * 16][0], 32);
      o[tn] = wmma_bf16(pa, vb, o[tn]);
    }
    if (ch + 1 < nchunks) lstoreKV(buf ^ 1);
  }

  // Normalize and store y[b][t][h*64 + d] as bf16
#pragma unroll
  for (int tn = 0; tn < 4; ++tn) {
#pragma unroll
    for (int v = 0; v < 8; ++v) {
      const int t = rowbase + v;
      const float val = o[tn][v] / lrow[v];
      Yb[((size_t)b * T_SZ + t) * C_SZ + h * HD + tn * 16 + ncol] = (bf16_t)val;
    }
  }
}

// ===========================================================================
// Kernel 3: out = y @ w_proj + b_proj.  M=4096, N=1024, K=1024.
// Same pipelined structure; A tile is already bf16 (16B copies).
// ===========================================================================
__global__ __launch_bounds__(256) void proj_gemm_kernel(
    const unsigned short* __restrict__ y_ws, const float* __restrict__ w_proj,
    const float* __restrict__ b_proj, float* __restrict__ out) {
  __shared__ __align__(16) bf16_t Xs[2][128][32];
  __shared__ __align__(16) bf16_t Ws[2][128][32];  // [n][k]

  const bf16_t* Yb = reinterpret_cast<const bf16_t*>(y_ws);

  const int tid  = threadIdx.x;
  const int wave = tid >> 5;
  const int lane = tid & 31;
  const int bN0  = blockIdx.x * 128;
  const int bM0  = blockIdx.y * 128;
  const int wM   = (wave >> 2) * 64;
  const int wN   = (wave & 3) * 32;

  v8f acc[4][2];
#pragma unroll
  for (int i = 0; i < 4; ++i)
#pragma unroll
    for (int j = 0; j < 2; ++j) acc[i][j] = {};

  v8bf yreg[2];
  float4 wreg[4];
  auto gload = [&](int k0) {
#pragma unroll
    for (int r = 0; r < 2; ++r) {
      const int i = tid + r * 256;              // 0..511 (128x32 / 8)
      const int m = i >> 2, k8 = (i & 3) << 3;
      yreg[r] = *reinterpret_cast<const v8bf*>(
          Yb + (size_t)(bM0 + m) * C_SZ + k0 + k8);
    }
#pragma unroll
    for (int r = 0; r < 4; ++r) {
      const int i = tid + r * 256;
      const int k = i >> 5, n4 = (i & 31) << 2;
      wreg[r] = *reinterpret_cast<const float4*>(
          w_proj + (size_t)(k0 + k) * C_SZ + bN0 + n4);
    }
  };
  auto lstore = [&](int buf) {
#pragma unroll
    for (int r = 0; r < 2; ++r) {
      const int i = tid + r * 256;
      const int m = i >> 2, k8 = (i & 3) << 3;
      *reinterpret_cast<v8bf*>(&Xs[buf][m][k8]) = yreg[r];
    }
#pragma unroll
    for (int r = 0; r < 4; ++r) {
      const int i = tid + r * 256;
      const int k = i >> 5, n4 = (i & 31) << 2;
      Ws[buf][n4 + 0][k] = (bf16_t)wreg[r].x;
      Ws[buf][n4 + 1][k] = (bf16_t)wreg[r].y;
      Ws[buf][n4 + 2][k] = (bf16_t)wreg[r].z;
      Ws[buf][n4 + 3][k] = (bf16_t)wreg[r].w;
    }
  };

  const int NKT = C_SZ / 32;
  gload(0);
  lstore(0);
  for (int kt = 0; kt < NKT; ++kt) {
    __syncthreads();
    const int buf = kt & 1;
    if (kt + 1 < NKT) gload((kt + 1) * 32);

    v16bf af[4];
#pragma unroll
    for (int tm = 0; tm < 4; ++tm)
      af[tm] = load_frag_a(&Xs[buf][wM + tm * 16][0], 32);
#pragma unroll
    for (int tn = 0; tn < 2; ++tn) {
      v16bf bfr = load_frag_b(&Ws[buf][wN + tn * 16][0], 32);
#pragma unroll
      for (int tm = 0; tm < 4; ++tm)
        acc[tm][tn] = wmma_bf16(af[tm], bfr, acc[tm][tn]);
    }
    if (kt + 1 < NKT) lstore(buf ^ 1);
  }

#pragma unroll
  for (int tm = 0; tm < 4; ++tm) {
#pragma unroll
    for (int tn = 0; tn < 2; ++tn) {
      const int n     = bN0 + wN + tn * 16 + (lane & 15);
      const int mbase = bM0 + wM + tm * 16 + ((lane >> 4) << 3);
      const float bias = b_proj[n];
#pragma unroll
      for (int v = 0; v < 8; ++v) {
        const int m = mbase + v;
        out[(size_t)m * C_SZ + n] = acc[tm][tn][v] + bias;
      }
    }
  }
}

// ===========================================================================
extern "C" void kernel_launch(void* const* d_in, const int* in_sizes, int n_in,
                              void* d_out, int out_size, void* d_ws, size_t ws_size,
                              hipStream_t stream) {
  (void)in_sizes; (void)n_in; (void)out_size; (void)ws_size;
  const float* x      = (const float*)d_in[0];
  const float* w_attn = (const float*)d_in[1];
  const float* b_attn = (const float*)d_in[2];
  const float* w_proj = (const float*)d_in[3];
  const float* b_proj = (const float*)d_in[4];
  // d_in[5] = n_heads (fixed to 16 here)

  // Workspace layout (bf16 elems): Q | K | V | Y  == 4 * 4194304 * 2B = 32 MB
  unsigned short* qkv = (unsigned short*)d_ws;
  unsigned short* yb  = qkv + (size_t)3 * HEAD_ELEMS;
  float* out = (float*)d_out;

  qkv_gemm_kernel<<<dim3(3 * C_SZ / 128, B_SZ * T_SZ / 128), 256, 0, stream>>>(
      x, w_attn, b_attn, qkv);
  attn_kernel<<<dim3(T_SZ / 64, B_SZ * NH), 128, 0, stream>>>(qkv, yb);
  proj_gemm_kernel<<<dim3(C_SZ / 128, B_SZ * T_SZ / 128), 256, 0, stream>>>(
      yb, w_proj, b_proj, out);
}
